// GraphTransformer_24644522344846
// MI455X (gfx1250) — matrix operand order repaired and checked by reading, not compiled, not run on hardware
//
#include <hip/hip_runtime.h>
#include <math.h>

// ---------------- problem constants ----------------
#define NN   20000          // nodes
#define EE   160000         // edges
#define INC0 256
#define HIDN 512
#define HEADS 8
#define DHEAD 64
#define NLAYERS 2

typedef __attribute__((ext_vector_type(16))) __bf16 v16bf;
typedef __attribute__((ext_vector_type(8)))  __bf16 v8bf;
typedef __attribute__((ext_vector_type(8)))  float  v8f;

// ---------------- helpers ----------------
__device__ __forceinline__ unsigned short f32_to_bf16_rne(float f) {
    unsigned int u = __float_as_uint(f);
    unsigned int r = (u + 0x7FFFu + ((u >> 16) & 1u)) >> 16;
    return (unsigned short)r;
}

__device__ __forceinline__ float gelu_tanh(float x) {
    float x3 = x * x * x;
    float t  = tanhf(0.7978845608028654f * (x + 0.044715f * x3));
    return 0.5f * x * (1.0f + t);
}

__device__ __forceinline__ void atomicMaxF(float* addr, float val) {
    unsigned int* ua  = (unsigned int*)addr;
    unsigned int  old = *ua;
    while (__uint_as_float(old) < val) {
        unsigned int assumed = old;
        old = atomicCAS(ua, assumed, __float_as_uint(val));
        if (old == assumed) break;
    }
}

// ---------------- conversion kernels ----------------
__global__ void f32_to_bf16_kernel(const float* __restrict__ in,
                                   unsigned short* __restrict__ out, int n) {
    int i = blockIdx.x * blockDim.x + threadIdx.x;
    if (i < n) out[i] = f32_to_bf16_rne(in[i]);
}

// W: [K, Nc] row-major f32  ->  Wt: [Nc, K] row-major bf16
__global__ void wt_transpose_bf16_kernel(const float* __restrict__ W,
                                         unsigned short* __restrict__ Wt,
                                         int K, int Nc) {
    int i = blockIdx.x * blockDim.x + threadIdx.x;
    if (i >= K * Nc) return;
    int k = i / Nc;
    int n = i - k * Nc;
    Wt[(size_t)n * K + k] = f32_to_bf16_rne(W[i]);
}

__global__ void init_accum_kernel(float* __restrict__ out,
                                  float* __restrict__ m,
                                  float* __restrict__ s,
                                  int n512, int n8) {
    int i = blockIdx.x * blockDim.x + threadIdx.x;
    if (i < n512) out[i] = 0.0f;
    if (i < n8) { m[i] = __uint_as_float(0xFF800000u); s[i] = 0.0f; }
}

__global__ void residual_add_kernel(const float* __restrict__ x,
                                    const float* __restrict__ h,
                                    float* __restrict__ out, int n) {
    int i = blockIdx.x * blockDim.x + threadIdx.x;
    if (i < n) out[i] = x[i] + h[i];
}

// ---------------- WMMA bf16 GEMM, 32x64 register-blocked per wave ----------------
// C[M,Nc](f32) = A[M,K](bf16) @ Bt[Nc,K](bf16, pre-transposed) + bias
// MODE: 0 = bias, 1 = bias+gelu(tanh), 2 = accumulate (C += A@B + bias)
// Each wave: 2 (M) x 4 (N) grid of 16x16 WMMA tiles -> 8 independent accumulators,
// so the 8 WMMAs per K-step have no RAW hazards between each other.
template <int MODE>
__global__ __launch_bounds__(128)
void gemm_bf16_wmma_kernel(const unsigned short* __restrict__ A,
                           const unsigned short* __restrict__ Bt,
                           const float* __restrict__ bias,
                           float* __restrict__ C,
                           int M, int K, int Nc) {
    int wave = threadIdx.x >> 5;
    int lane = threadIdx.x & 31;
    int tilesN = Nc >> 6;                       // 64-wide wave tiles
    int tile = blockIdx.x * 4 + wave;
    int tilesTotal = (M >> 5) * tilesN;         // 32-tall wave tiles
    if (tile >= tilesTotal) return;
    int tm = tile / tilesN;
    int tn = tile - tm * tilesN;

    int r = lane & 15;   // A row within subtile / B column within subtile
    int g = lane >> 4;   // lane half selector

    const unsigned short* Arow0 = A + (size_t)(tm * 32 + r) * K;
    const unsigned short* Arow1 = Arow0 + (size_t)16 * K;
    const unsigned short* Brow0 = Bt + (size_t)(tn * 64 + r) * K;
    const unsigned short* Brow1 = Brow0 + (size_t)16 * K;
    const unsigned short* Brow2 = Brow0 + (size_t)32 * K;
    const unsigned short* Brow3 = Brow0 + (size_t)48 * K;

    v8f acc[2][4] = {};

    for (int k0 = 0; k0 < K; k0 += 32) {
        // A fragments: lane holds 16 contiguous K starting at g*16 (matches B)
        v16bf a[2];
        {
            v8bf lo0 = *(const v8bf*)(Arow0 + k0 + g * 16);
            v8bf hi0 = *(const v8bf*)(Arow0 + k0 + g * 16 + 8);
            v8bf lo1 = *(const v8bf*)(Arow1 + k0 + g * 16);
            v8bf hi1 = *(const v8bf*)(Arow1 + k0 + g * 16 + 8);
            #pragma unroll
            for (int i = 0; i < 8; i++) {
                a[0][i] = lo0[i]; a[0][8 + i] = hi0[i];
                a[1][i] = lo1[i]; a[1][8 + i] = hi1[i];
            }
        }
        // B fragments: lane holds 16 contiguous K starting at g*16 (same order as A)
        v16bf b[4];
        b[0] = *(const v16bf*)(Brow0 + k0 + g * 16);
        b[1] = *(const v16bf*)(Brow1 + k0 + g * 16);
        b[2] = *(const v16bf*)(Brow2 + k0 + g * 16);
        b[3] = *(const v16bf*)(Brow3 + k0 + g * 16);

        #pragma unroll
        for (int i = 0; i < 2; i++) {
            #pragma unroll
            for (int j = 0; j < 4; j++) {
                acc[i][j] = __builtin_amdgcn_wmma_f32_16x16x32_bf16(
                    false, a[i], false, b[j], (short)0, acc[i][j], false, false);
            }
        }
    }

    // branch-free epilogue: bias hoisted, MODE resolved at compile time
    float bv[4];
    #pragma unroll
    for (int j = 0; j < 4; j++) bv[j] = bias[tn * 64 + j * 16 + r];

    #pragma unroll
    for (int j = 0; j < 4; j++) {
        int cn = tn * 64 + j * 16 + r;
        #pragma unroll
        for (int i = 0; i < 2; i++) {
            #pragma unroll
            for (int rr = 0; rr < 8; rr++) {
                int cm = tm * 32 + i * 16 + rr + 8 * g;   // C layout: VGPR rr -> M = rr + 8*g
                size_t idx = (size_t)cm * Nc + cn;
                float val = acc[i][j][rr] + bv[j];
                if (MODE == 1)      val = gelu_tanh(val);
                else if (MODE == 2) val += C[idx];
                C[idx] = val;
            }
        }
    }
}

// ---------------- edge kernels (one wave32 per edge) ----------------
// alpha[e,h] = sum_d q[dst][h,d]*(k[src][h,d]+e[h,d]) / sqrt(64); atomic max into m
__global__ void edge_alpha_kernel(const float* __restrict__ q,
                                  const float* __restrict__ k,
                                  const float* __restrict__ ea,
                                  const float* __restrict__ We,   // [3,512] f32
                                  const int* __restrict__ src,
                                  const int* __restrict__ dst,
                                  float* __restrict__ alpha,
                                  float* __restrict__ m, int E) {
    int e    = (blockIdx.x * blockDim.x + threadIdx.x) >> 5;
    int lane = threadIdx.x & 31;
    if (e >= E) return;
    int sN = src[e], dN = dst[e];
    float a0 = ea[e * 3 + 0], a1 = ea[e * 3 + 1], a2 = ea[e * 3 + 2];
    const float* qd = q + (size_t)dN * HIDN;
    const float* ks = k + (size_t)sN * HIDN;
    float part = 0.0f;
    int c0 = lane * 16;
    #pragma unroll
    for (int i = 0; i < 16; i++) {
        int c = c0 + i;
        float ec = a0 * We[c] + a1 * We[512 + c] + a2 * We[1024 + c];
        part += qd[c] * (ks[c] + ec);
    }
    part += __shfl_xor(part, 1, 32);   // reduce 4 lanes per head (64 ch / 16 per lane)
    part += __shfl_xor(part, 2, 32);
    if ((lane & 3) == 0) {
        int h = lane >> 2;
        float val = part * 0.125f;     // 1/sqrt(64)
        alpha[(size_t)e * HEADS + h] = val;
        atomicMaxF(&m[(size_t)dN * HEADS + h], val);
    }
}

// a = exp(alpha - m[dst]); atomic-add into s; alpha buffer overwritten with a
__global__ void edge_exp_kernel(float* __restrict__ alpha,
                                const float* __restrict__ m,
                                float* __restrict__ s,
                                const int* __restrict__ dst, int E) {
    int i = blockIdx.x * blockDim.x + threadIdx.x;
    if (i >= E * HEADS) return;
    int e = i >> 3, h = i & 7;
    int dN = dst[e];
    float a = __expf(alpha[i] - m[(size_t)dN * HEADS + h]);
    alpha[i] = a;
    atomicAdd(&s[(size_t)dN * HEADS + h], a);
}

// out[dst] += (a/s[dst]) * (v[src]+e)
__global__ void edge_scatter_kernel(const float* __restrict__ a,
                                    const float* __restrict__ ssum,
                                    const float* __restrict__ v,
                                    const float* __restrict__ ea,
                                    const float* __restrict__ We,
                                    const int* __restrict__ src,
                                    const int* __restrict__ dst,
                                    float* __restrict__ out, int E) {
    int e    = (blockIdx.x * blockDim.x + threadIdx.x) >> 5;
    int lane = threadIdx.x & 31;
    if (e >= E) return;
    int sN = src[e], dN = dst[e];
    int h = lane >> 2;
    float w = a[(size_t)e * HEADS + h] / ssum[(size_t)dN * HEADS + h];
    float a0 = ea[e * 3 + 0], a1 = ea[e * 3 + 1], a2 = ea[e * 3 + 2];
    const float* vs = v + (size_t)sN * HIDN;
    float* od = out + (size_t)dN * HIDN;
    int c0 = lane * 16;
    #pragma unroll
    for (int i = 0; i < 16; i++) {
        int c = c0 + i;
        float ec = a0 * We[c] + a1 * We[512 + c] + a2 * We[1024 + c];
        atomicAdd(&od[c], w * (vs[c] + ec));
    }
}

// ---------------- host launch ----------------
extern "C" void kernel_launch(void* const* d_in, const int* in_sizes, int n_in,
                              void* d_out, int out_size, void* d_ws, size_t ws_size,
                              hipStream_t stream) {
    (void)in_sizes; (void)n_in; (void)out_size; (void)ws_size;

    const float* x0  = (const float*)d_in[0];           // [N, 256]
    const int*   ei  = (const int*)d_in[1];             // [2, E]
    const float* ea  = (const float*)d_in[2];           // [E, 3]
    const int* src = ei;
    const int* dst = ei + EE;

    // ---- workspace layout (bytes, all 64B aligned) ----
    char* ws = (char*)d_ws;
    unsigned short* Abf = (unsigned short*)(ws + 0);              // N*512 bf16  (20.48 MB)
    float* qf    = (float*)(ws + 20480000);                       // N*512 f32
    float* kf    = (float*)(ws + 61440000);
    float* vf    = (float*)(ws + 102400000);
    float* of    = (float*)(ws + 143360000);
    float* alpha = (float*)(ws + 184320000);                      // E*8 f32
    float* mbuf  = (float*)(ws + 189440000);                      // N*8 f32
    float* sbuf  = (float*)(ws + 190080000);                      // N*8 f32
    unsigned short* wts = (unsigned short*)(ws + 190720000);      // 6 * 512*512 bf16
    const size_t WSLOT = 512 * 512;

    float* xout = (float*)d_out;   // inter-layer x buffer + final output

    const float* xin = x0;
    int inC = INC0;

    for (int l = 0; l < NLAYERS; l++) {
        const float* Wq    = (const float*)d_in[3 + 13 * l + 0];
        const float* bq    = (const float*)d_in[3 + 13 * l + 1];
        const float* Wk    = (const float*)d_in[3 + 13 * l + 2];
        const float* bk    = (const float*)d_in[3 + 13 * l + 3];
        const float* Wv    = (const float*)d_in[3 + 13 * l + 4];
        const float* bv    = (const float*)d_in[3 + 13 * l + 5];
        const float* We    = (const float*)d_in[3 + 13 * l + 6];
        const float* Wskip = (const float*)d_in[3 + 13 * l + 7];
        const float* bskip = (const float*)d_in[3 + 13 * l + 8];
        const float* W1    = (const float*)d_in[3 + 13 * l + 9];
        const float* b1    = (const float*)d_in[3 + 13 * l + 10];
        const float* W2    = (const float*)d_in[3 + 13 * l + 11];
        const float* b2    = (const float*)d_in[3 + 13 * l + 12];

        unsigned short* WqT = wts + 0 * WSLOT;
        unsigned short* WkT = wts + 1 * WSLOT;
        unsigned short* WvT = wts + 2 * WSLOT;
        unsigned short* WsT = wts + 3 * WSLOT;
        unsigned short* W1T = wts + 4 * WSLOT;
        unsigned short* W2T = wts + 5 * WSLOT;

        int wElems = inC * HIDN, wBlocks = (wElems + 255) / 256;
        wt_transpose_bf16_kernel<<<wBlocks, 256, 0, stream>>>(Wq, WqT, inC, HIDN);
        wt_transpose_bf16_kernel<<<wBlocks, 256, 0, stream>>>(Wk, WkT, inC, HIDN);
        wt_transpose_bf16_kernel<<<wBlocks, 256, 0, stream>>>(Wv, WvT, inC, HIDN);
        wt_transpose_bf16_kernel<<<wBlocks, 256, 0, stream>>>(Wskip, WsT, inC, HIDN);
        int w2Blocks = (HIDN * HIDN + 255) / 256;
        wt_transpose_bf16_kernel<<<w2Blocks, 256, 0, stream>>>(W1, W1T, HIDN, HIDN);
        wt_transpose_bf16_kernel<<<w2Blocks, 256, 0, stream>>>(W2, W2T, HIDN, HIDN);

        // x -> bf16
        int xn = NN * inC;
        f32_to_bf16_kernel<<<(xn + 255) / 256, 256, 0, stream>>>(xin, Abf, xn);

        // q/k/v projections: wave tiles are 32x64 -> (M/32)*(N/64) tiles, 4 waves/block
        const int waveTiles = (NN / 32) * (HIDN / 64);     // 625 * 8 = 5000
        const int gBlocks   = (waveTiles + 3) / 4;         // 1250
        gemm_bf16_wmma_kernel<0><<<gBlocks, 128, 0, stream>>>(Abf, WqT, bq, qf, NN, inC, HIDN);
        gemm_bf16_wmma_kernel<0><<<gBlocks, 128, 0, stream>>>(Abf, WkT, bk, kf, NN, inC, HIDN);
        gemm_bf16_wmma_kernel<0><<<gBlocks, 128, 0, stream>>>(Abf, WvT, bv, vf, NN, inC, HIDN);

        // init attention accumulators
        init_accum_kernel<<<(NN * HIDN + 255) / 256, 256, 0, stream>>>(of, mbuf, sbuf, NN * HIDN, NN * HEADS);

        // edge phase (wave per edge)
        edge_alpha_kernel<<<EE / 8, 256, 0, stream>>>(qf, kf, ea, We, src, dst, alpha, mbuf, EE);
        edge_exp_kernel<<<(EE * HEADS + 255) / 256, 256, 0, stream>>>(alpha, mbuf, sbuf, dst, EE);
        edge_scatter_kernel<<<EE / 8, 256, 0, stream>>>(alpha, sbuf, vf, ea, We, src, dst, of, EE);

        // skip: of += x @ Wskip + bskip   (accumulate mode)
        gemm_bf16_wmma_kernel<2><<<gBlocks, 128, 0, stream>>>(Abf, WsT, bskip, of, NN, inC, HIDN);

        // MLP: h1 = gelu(of@W1+b1); h2 = gelu(h1@W2+b2); x_new = of + h2
        int hn = NN * HIDN;
        f32_to_bf16_kernel<<<(hn + 255) / 256, 256, 0, stream>>>(of, Abf, hn);
        gemm_bf16_wmma_kernel<1><<<gBlocks, 128, 0, stream>>>(Abf, W1T, b1, qf, NN, HIDN, HIDN);
        f32_to_bf16_kernel<<<(hn + 255) / 256, 256, 0, stream>>>(qf, Abf, hn);
        gemm_bf16_wmma_kernel<1><<<gBlocks, 128, 0, stream>>>(Abf, W2T, b2, kf, NN, HIDN, HIDN);
        residual_add_kernel<<<(hn + 255) / 256, 256, 0, stream>>>(of, kf, xout, hn);

        xin = xout;
        inC = HIDN;
    }
}
